// DeltaRule_20194936225992
// MI455X (gfx1250) — matrix operand order repaired and checked by reading, not compiled, not run on hardware
//
#include <hip/hip_runtime.h>
#include <stdint.h>

// ---------------------------------------------------------------------------
// DeltaRule (diagonal-state linear attention) for MI455X / gfx1250, wave32.
//   * bf16 WMMA (v_wmma_f32_16x16x32_bf16) tiled GEMMs, double-buffered LDS,
//     register-prefetched global loads, v_perm-based f32->bf16 packing
//   * linear recurrence rewritten as chunked parallel scan (3 passes)
// ---------------------------------------------------------------------------

typedef __bf16 bf16;
typedef bf16  v16bf __attribute__((ext_vector_type(16)));
typedef bf16  v8bf  __attribute__((ext_vector_type(8)));
typedef float v8f   __attribute__((ext_vector_type(8)));

#define H_DIM    1024
#define HEAD_DIM 64
#define N_HEADS  16
#define BATCH    4
#define SEQ      2048
#define M_TOTAL  (BATCH * SEQ)   // 8192 rows
#define K_DIM    H_DIM           // 1024 reduction dim

#define BLK_M 128
#define BLK_N 64
#define BLK_K 32
#define K_ITERS (K_DIM / BLK_K)  // 32
#define LDSW  20                 // u32 per LDS row (32 bf16 data + 8 bf16 pad)

#define N_CHANNELS (BATCH * H_DIM)   // 4096 (b, h, d) diagonal channels
#define CHUNK      64
#define N_CHUNKS   (SEQ / CHUNK)     // 32

// Pack two f32 into packed bf16 (round-half-up): 2 adds + 1 v_perm_b32.
__device__ __forceinline__ uint32_t pack2_bf16(float a, float b) {
    uint32_t ua = __float_as_uint(a) + 0x8000u;
    uint32_t ub = __float_as_uint(b) + 0x8000u;
    // dst = { ub[31:16], ua[31:16] } : bytes {7,6} of src0=ub, {3,2} of src1=ua
    return __builtin_amdgcn_perm(ub, ua, 0x07060302u);
}

// MODE: 0 = phi (elu+1), 1 = identity, 2 = sigmoid
// Y[m, n] = act( sum_k X[m,k] * W[n,k] + bias[n] )
template <int MODE>
__global__ __launch_bounds__(256)
void proj_gemm(const float* __restrict__ X, const float* __restrict__ W,
               const float* __restrict__ bias, float* __restrict__ Y, int Nout)
{
    __shared__ __align__(16) uint32_t lA[2][BLK_M * LDSW];
    __shared__ __align__(16) uint32_t lB[2][BLK_N * LDSW];

    const int tid  = threadIdx.x;
    const int lane = tid & 31;
    const int wid  = tid >> 5;           // 8 waves
    const int half = lane >> 4;          // K-half selector (ISA 16-bit A layout)
    const int r    = lane & 15;          // row (A) / column (B/C/D) within tile
    const int mWave = (wid & 3) * 32;    // 4 wave-rows
    const int nWave = (wid >> 2) * 32;   // 2 wave-cols
    const int mBlock = blockIdx.x * BLK_M;
    const int nBlock = blockIdx.y * BLK_N;

    const int srow = tid >> 2;           // 0..63 staging row
    const int scol = (tid & 3) * 8;      // 0,8,16,24 staging col (k)

    v8f acc[2][2] = {};

    // -------- register prefetch of one global K-tile --------
    float4 ra[2][2];                     // A: two 64-row passes x 8 floats
    float4 rb[2];                        // B: 8 floats
    auto gload = [&](int k0) {
        #pragma unroll
        for (int p = 0; p < 2; ++p) {
            const float* src =
                X + (size_t)(mBlock + srow + p * 64) * K_DIM + k0 + scol;
            ra[p][0] = *(const float4*)src;
            ra[p][1] = *(const float4*)(src + 4);
        }
        const int n = nBlock + srow;
        if (n < Nout) {
            const float* src = W + (size_t)n * K_DIM + k0 + scol;
            rb[0] = *(const float4*)src;
            rb[1] = *(const float4*)(src + 4);
        } else {
            rb[0] = make_float4(0.f, 0.f, 0.f, 0.f);
            rb[1] = make_float4(0.f, 0.f, 0.f, 0.f);
        }
    };

    gload(0);
    int buf = 0;
    for (int it = 0; it < K_ITERS; ++it) {
        // ---- stage prefetched tile into LDS (f32 -> bf16) ----
        #pragma unroll
        for (int p = 0; p < 2; ++p) {
            uint4 pk;
            pk.x = pack2_bf16(ra[p][0].x, ra[p][0].y);
            pk.y = pack2_bf16(ra[p][0].z, ra[p][0].w);
            pk.z = pack2_bf16(ra[p][1].x, ra[p][1].y);
            pk.w = pack2_bf16(ra[p][1].z, ra[p][1].w);
            *(uint4*)&lA[buf][(srow + p * 64) * LDSW + (scol >> 1)] = pk;
        }
        {
            uint4 pk;
            pk.x = pack2_bf16(rb[0].x, rb[0].y);
            pk.y = pack2_bf16(rb[0].z, rb[0].w);
            pk.z = pack2_bf16(rb[1].x, rb[1].y);
            pk.w = pack2_bf16(rb[1].z, rb[1].w);
            *(uint4*)&lB[buf][srow * LDSW + (scol >> 1)] = pk;
        }
        __syncthreads();   // single barrier per K-step (double-buffered LDS)

        // ---- issue next tile's global loads early ----
        if (it + 1 < K_ITERS) gload((it + 1) * BLK_K);

        // ---- fragments per CDNA5 WMMA VGPR layouts ----
        v16bf afrag[2], bfrag[2];
        #pragma unroll
        for (int i = 0; i < 2; ++i) {
            // A 16x32: lane half 0 -> K{0..7,16..23}, half 1 -> K{8..15,24..31}
            const uint32_t* p0 =
                &lA[buf][(mWave + i * 16 + r) * LDSW + half * 4];
            v8bf lo = *(const v8bf*)p0;
            v8bf hi = *(const v8bf*)(p0 + 8);
            afrag[i] = __builtin_shufflevector(lo, hi, 0,1,2,3,4,5,6,7,
                                                       8,9,10,11,12,13,14,15);
        }
        #pragma unroll
        for (int j = 0; j < 2; ++j) {
            // B 32x16: lane = column, half 0 -> K{0..7,16..23}, half 1 -> K{8..15,24..31}
            const uint32_t* p0 =
                &lB[buf][(nWave + j * 16 + r) * LDSW + half * 4];
            v8bf lo = *(const v8bf*)p0;
            v8bf hi = *(const v8bf*)(p0 + 8);
            bfrag[j] = __builtin_shufflevector(lo, hi, 0,1,2,3,4,5,6,7,
                                                       8,9,10,11,12,13,14,15);
        }
        #pragma unroll
        for (int i = 0; i < 2; ++i)
            #pragma unroll
            for (int j = 0; j < 2; ++j)
                acc[i][j] = __builtin_amdgcn_wmma_f32_16x16x32_bf16(
                    false, afrag[i], false, bfrag[j],
                    (short)0, acc[i][j], false, false);

        buf ^= 1;
    }

    // ---- epilogue: D layout VGPR i -> M = 8*half + i, N = lane&15 ----
    #pragma unroll
    for (int j = 0; j < 2; ++j) {
        const int n = nBlock + nWave + j * 16 + r;
        if (n >= Nout) continue;
        const float bn = bias[n];
        #pragma unroll
        for (int i = 0; i < 2; ++i) {
            const int mBase = mBlock + mWave + i * 16 + half * 8;
            #pragma unroll
            for (int vi = 0; vi < 8; ++vi) {
                float x = acc[i][j][vi] + bn;
                if (MODE == 0)      x = (x > 0.f) ? x + 1.f : __expf(x); // elu+1
                else if (MODE == 2) x = 1.f / (1.f + __expf(-x));       // sigmoid
                Y[(size_t)(mBase + vi) * Nout + n] = x;
            }
        }
    }
}

// ---- scan pass 1: per (channel, chunk) compose affine maps  s' = a*s + c ----
// a_t = 1 - b_t*pk_t^2 ; c_t = b_t*v_t*pk_t
__global__ __launch_bounds__(256)
void scan_chunk_reduce(const float* __restrict__ pk, const float* __restrict__ v,
                       const float* __restrict__ bg,
                       float* __restrict__ Aout, float* __restrict__ Cout)
{
    const int idx     = blockIdx.x * 256 + threadIdx.x;  // chunk*4096 + channel
    const int channel = idx & (N_CHANNELS - 1);
    const int chunk   = idx >> 12;
    const int b  = channel >> 10;
    const int hd = channel & (H_DIM - 1);
    const int h  = hd >> 6;
    const float* pkp = pk + (size_t)b * SEQ * H_DIM + hd;
    const float* vp  = v  + (size_t)b * SEQ * H_DIM + hd;
    const float* bp  = bg + (size_t)b * SEQ * N_HEADS + h;
    const int t0 = chunk * CHUNK;

    float A = 1.f, C = 0.f;
    #pragma unroll 4
    for (int i = 0; i < CHUNK; ++i) {
        const int t = t0 + i;
        const float pkt = pkp[(size_t)t * H_DIM];
        const float vt  = vp [(size_t)t * H_DIM];
        const float bt  = bp [(size_t)t * N_HEADS];
        const float a = 1.f - bt * pkt * pkt;
        const float c = bt * vt * pkt;
        A = a * A;
        C = a * C + c;
    }
    Aout[idx] = A;
    Cout[idx] = C;
}

// ---- scan pass 2: tiny serial prefix over 32 chunks per channel ----
__global__ __launch_bounds__(256)
void scan_prefix(const float* __restrict__ A, const float* __restrict__ C,
                 float* __restrict__ S0)
{
    const int channel = blockIdx.x * 256 + threadIdx.x;  // 4096 channels
    float s = 0.f;
    #pragma unroll
    for (int ch = 0; ch < N_CHUNKS; ++ch) {
        const int idx = ch * N_CHANNELS + channel;
        S0[idx] = s;                  // state entering chunk ch
        s = A[idx] * s + C[idx];
    }
}

// ---- scan pass 3: replay recurrence inside each chunk, emit y = s*pq ----
__global__ __launch_bounds__(256)
void scan_apply(const float* __restrict__ pk, const float* __restrict__ pq,
                const float* __restrict__ v,  const float* __restrict__ bg,
                const float* __restrict__ S0, float* __restrict__ y)
{
    const int idx     = blockIdx.x * 256 + threadIdx.x;
    const int channel = idx & (N_CHANNELS - 1);
    const int chunk   = idx >> 12;
    const int b  = channel >> 10;
    const int hd = channel & (H_DIM - 1);
    const int h  = hd >> 6;
    const size_t base = (size_t)b * SEQ * H_DIM + hd;
    const float* pkp = pk + base;
    const float* pqp = pq + base;
    const float* vp  = v  + base;
    const float* bp  = bg + (size_t)b * SEQ * N_HEADS + h;
    float* yp = y + base;
    const int t0 = chunk * CHUNK;

    float s = S0[idx];
    #pragma unroll 4
    for (int i = 0; i < CHUNK; ++i) {
        const int t = t0 + i;
        const float pkt = pkp[(size_t)t * H_DIM];
        const float pqt = pqp[(size_t)t * H_DIM];
        const float vt  = vp [(size_t)t * H_DIM];
        const float bt  = bp [(size_t)t * N_HEADS];
        s = s + bt * (vt - s * pkt) * pkt;
        yp[(size_t)t * H_DIM] = s * pqt;
    }
}

extern "C" void kernel_launch(void* const* d_in, const int* in_sizes, int n_in,
                              void* d_out, int out_size, void* d_ws, size_t ws_size,
                              hipStream_t stream)
{
    (void)in_sizes; (void)n_in; (void)out_size; (void)ws_size;
    const float* query = (const float*)d_in[0];
    const float* key_  = (const float*)d_in[1];
    const float* value = (const float*)d_in[2];
    const float* beta  = (const float*)d_in[3];
    const float* Wq = (const float*)d_in[4];
    const float* bq = (const float*)d_in[5];
    const float* Wk = (const float*)d_in[6];
    const float* bk = (const float*)d_in[7];
    const float* Wv = (const float*)d_in[8];
    const float* bv = (const float*)d_in[9];
    const float* Wb = (const float*)d_in[10];
    const float* bb = (const float*)d_in[11];
    const float* Wo = (const float*)d_in[12];
    const float* bo = (const float*)d_in[13];
    float* out = (float*)d_out;

    const size_t MK = (size_t)M_TOTAL * H_DIM;              // 8M floats
    float* q  = (float*)d_ws;                               // phi(q)   32 MB
    float* k  = q  + MK;                                    // phi(k)   32 MB
    float* v  = k  + MK;                                    // v        32 MB
    float* bs = v  + MK;                                    // gate     0.5 MB
    float* y  = bs + (size_t)M_TOTAL * N_HEADS;             // y        32 MB
    float* Ac = y  + MK;                                    // chunk A
    float* Cc = Ac + (size_t)N_CHANNELS * N_CHUNKS;         // chunk C
    float* S0 = Cc + (size_t)N_CHANNELS * N_CHUNKS;         // chunk entry states

    const dim3 blk(256);
    const dim3 gFull(M_TOTAL / BLK_M, H_DIM / BLK_N);       // (64, 16)
    const dim3 gBeta(M_TOTAL / BLK_M, 1);
    const int  gScan = (N_CHANNELS * N_CHUNKS) / 256;       // 512

    proj_gemm<0><<<gFull, blk, 0, stream>>>(query, Wq, bq, q, H_DIM);   // phi(q)
    proj_gemm<0><<<gFull, blk, 0, stream>>>(key_,  Wk, bk, k, H_DIM);   // phi(k)
    proj_gemm<1><<<gFull, blk, 0, stream>>>(value, Wv, bv, v, H_DIM);   // v
    proj_gemm<2><<<gBeta, blk, 0, stream>>>(beta,  Wb, bb, bs, N_HEADS);// sigmoid gate

    scan_chunk_reduce<<<gScan, blk, 0, stream>>>(k, v, bs, Ac, Cc);
    scan_prefix<<<N_CHANNELS / 256, blk, 0, stream>>>(Ac, Cc, S0);
    scan_apply<<<gScan, blk, 0, stream>>>(k, q, v, bs, S0, y);

    proj_gemm<1><<<gFull, blk, 0, stream>>>(y, Wo, bo, out, H_DIM);     // output proj
}
